// DifferentialCausalSelfAttention_48859547959913
// MI455X (gfx1250) — matrix operand order repaired and checked by reading, not compiled, not run on hardware
//
#include <hip/hip_runtime.h>
#include <stdint.h>

#define BATCH 4
#define SEQ   2048
#define EMB   1024
#define NH    8
#define HD    64
#define LAMBDA_INIT 0.6192834727f   /* 0.8 - 0.6*exp(-0.3*4) */
#define RMS_EPS 1.1920929e-7f
#define LN_EPS  1e-5f

typedef _Float16 f16;
typedef __attribute__((ext_vector_type(16))) _Float16 v16h;
typedef __attribute__((ext_vector_type(8)))  float    v8f;

union H16 { v16h v; uint4 u[2]; };

static __device__ __forceinline__ v16h load16(const f16* p) {
  H16 r; const uint4* q = (const uint4*)p; r.u[0] = q[0]; r.u[1] = q[1]; return r.v;
}
// A-fragment (16x32 f16, row-major source): lane row = l&15, K = hi*16+{0..15} contiguous
static __device__ __forceinline__ v16h load_afrag(const f16* row, int k0, int hi) {
  H16 r;
  r.u[0] = *(const uint4*)(row + k0 + hi * 16);
  r.u[1] = *(const uint4*)(row + k0 + hi * 16 + 8);
  return r.v;
}
static __device__ __forceinline__ v8f wmma16(v16h a, v16h b, v8f c) {
  return __builtin_amdgcn_wmma_f32_16x16x32_f16(false, a, false, b, (short)0, c, false, false);
}

// ---------------- cast fp32 -> fp16 ----------------
__global__ void k_cast_h(const float* __restrict__ src, f16* __restrict__ dst, int n) {
  int i = blockIdx.x * blockDim.x + threadIdx.x;
  if (i < n) dst[i] = (f16)src[i];
}

// ---------------- GEMM: Y[M,N] = A[M,K] @ W[N,K]^T, f16 in / f32 out ----------------
__global__ void __launch_bounds__(256) k_gemm_xwT(const f16* __restrict__ A,
                                                  const f16* __restrict__ W,
                                                  float* __restrict__ Y,
                                                  int M, int N, int K) {
  const int lane = threadIdx.x & 31;
  const int wv   = threadIdx.x >> 5;
  const int ln   = lane & 15;
  const int hi   = lane >> 4;
  const int m0 = blockIdx.y * 256 + wv * 32;   // wave tile 32x64
  const int n0 = blockIdx.x * 64;
  v8f acc[2][4] = {};
  const f16* arow0 = A + (size_t)(m0 + ln) * K;
  const f16* arow1 = A + (size_t)(m0 + 16 + ln) * K;
  for (int k0 = 0; k0 < K; k0 += 32) {
    v16h a0 = load_afrag(arow0, k0, hi);
    v16h a1 = load_afrag(arow1, k0, hi);
#pragma unroll
    for (int ni = 0; ni < 4; ++ni) {
      // B[k,n] = W[n,k]; lane col = ln, K = hi*16 + {0..15} contiguous
      v16h bf = load16(W + (size_t)(n0 + ni * 16 + ln) * K + k0 + hi * 16);
      acc[0][ni] = wmma16(a0, bf, acc[0][ni]);
      acc[1][ni] = wmma16(a1, bf, acc[1][ni]);
    }
  }
#pragma unroll
  for (int mi = 0; mi < 2; ++mi)
#pragma unroll
    for (int ni = 0; ni < 4; ++ni)
#pragma unroll
      for (int i = 0; i < 8; ++i)
        Y[(size_t)(m0 + mi * 16 + hi * 8 + i) * N + n0 + ni * 16 + ln] = acc[mi][ni][i];
}

// ---------------- rmsnorm + RoPE (+scale) -> (b,sh,T,64) f16 ----------------
__global__ void k_prep_qk(const float* __restrict__ raw, f16* __restrict__ out, float scale) {
  const int lane = threadIdx.x & 31;
  const int wv   = threadIdx.x >> 5;
  const int r = blockIdx.x * 8 + wv;          // r = (b*T + t)*16 + sh
  const int sh = r & 15;
  const int bt = r >> 4;
  const int t  = bt & (SEQ - 1);
  const int b  = bt >> 11;
  float x1 = raw[(size_t)r * 64 + lane];
  float x2 = raw[(size_t)r * 64 + 32 + lane];
  float ss = x1 * x1 + x2 * x2;
  for (int m = 16; m >= 1; m >>= 1) ss += __shfl_xor(ss, m, 32);
  float rn = rsqrtf(ss * (1.0f / 64.0f) + RMS_EPS);
  // inv_freq[d] = 10000^(-2d/64) = exp(-ln(10000)/32 * d)
  float fr = (float)t * __expf(-0.28782313662f * (float)lane);
  float c = cosf(fr), s = sinf(fr);
  float n1 = x1 * rn, n2 = x2 * rn;
  float y1 = (n1 * c + n2 * s) * scale;
  float y2 = (n2 * c - n1 * s) * scale;
  size_t o = (((size_t)(b * 16 + sh)) * SEQ + t) * 64;
  out[o + lane]      = (f16)y1;
  out[o + 32 + lane] = (f16)y2;
}

// ---------------- v transpose: (b,t,h,128) f32 -> (b,h,128,T) f16 ----------------
__global__ void k_prep_v(const float* __restrict__ raw, f16* __restrict__ vT) {
  int i = blockIdx.x * blockDim.x + threadIdx.x;   // ((b*8+h)*128 + d)*T + t
  int t = i & (SEQ - 1);
  int d = (i >> 11) & 127;
  int h = (i >> 18) & 7;
  int b = i >> 21;
  vT[i] = (f16)raw[(((size_t)(b * SEQ + t)) * 8 + h) * 128 + d];
}

// ---------------- lambda scalar ----------------
__global__ void k_lambda(const float* lq1, const float* lk1,
                         const float* lq2, const float* lk2, float* lam) {
  int l = threadIdx.x;
  float s1 = lq1[l] * lk1[l] + lq1[l + 32] * lk1[l + 32];
  float s2 = lq2[l] * lk2[l] + lq2[l + 32] * lk2[l + 32];
  for (int m = 16; m >= 1; m >>= 1) { s1 += __shfl_xor(s1, m, 32); s2 += __shfl_xor(s2, m, 32); }
  if (l == 0) lam[0] = __expf(s1) - __expf(s2) + LAMBDA_INIT;
}

// ---------------- flash attention per sub-head -> O (b,sh,T,128) f32 ----------------
__global__ void __launch_bounds__(256) k_flash(const f16* __restrict__ qrot,
                                               const f16* __restrict__ krot,
                                               const f16* __restrict__ vT,
                                               float* __restrict__ Obuf) {
  __shared__ __align__(16) f16 tile[8][16][32];   // per-wave P staging
  const int lane = threadIdx.x & 31;
  const int wv   = threadIdx.x >> 5;
  const int ln   = lane & 15;
  const int hi   = lane >> 4;
  const int sh = blockIdx.y & 15;
  const int b  = blockIdx.y >> 4;
  const int h  = sh >> 1;
  const int q0 = blockIdx.x * 128 + wv * 16;
  const f16* qbase = qrot + ((size_t)(b * 16 + sh)) * SEQ * 64;
  const f16* kbase = krot + ((size_t)(b * 16 + sh)) * SEQ * 64;
  const f16* vbase = vT   + ((size_t)(b * 8 + h)) * 128 * SEQ;

  v16h qa0 = load_afrag(qbase + (size_t)(q0 + ln) * 64, 0, hi);
  v16h qa1 = load_afrag(qbase + (size_t)(q0 + ln) * 64, 32, hi);

  v8f O[8] = {};
  float mrow[8], srow[8];
#pragma unroll
  for (int i = 0; i < 8; ++i) { mrow[i] = -1e30f; srow[i] = 0.0f; }

  for (int n0 = 0; n0 < q0 + 16; n0 += 32) {
    v8f s0 = {}, s1v = {};
    s0  = wmma16(qa0, load16(kbase + (size_t)(n0 + ln) * 64 + hi * 16), s0);
    s0  = wmma16(qa1, load16(kbase + (size_t)(n0 + ln) * 64 + 32 + hi * 16), s0);
    s1v = wmma16(qa0, load16(kbase + (size_t)(n0 + 16 + ln) * 64 + hi * 16), s1v);
    s1v = wmma16(qa1, load16(kbase + (size_t)(n0 + 16 + ln) * 64 + 32 + hi * 16), s1v);
    const bool full = (n0 + 31 <= q0);
#pragma unroll
    for (int i = 0; i < 8; ++i) {
      const int row = q0 + hi * 8 + i;
      float e0 = s0[i], e1 = s1v[i];
      if (!full) {
        if (n0 + ln > row)      e0 = -1e30f;
        if (n0 + 16 + ln > row) e1 = -1e30f;
      }
      float tm = fmaxf(e0, e1);
      tm = fmaxf(tm, __shfl_xor(tm, 1, 32));
      tm = fmaxf(tm, __shfl_xor(tm, 2, 32));
      tm = fmaxf(tm, __shfl_xor(tm, 4, 32));
      tm = fmaxf(tm, __shfl_xor(tm, 8, 32));
      float nm = fmaxf(mrow[i], tm);
      float alpha = __expf(mrow[i] - nm);
      mrow[i] = nm;
      float p0 = __expf(e0 - nm);
      float p1 = __expf(e1 - nm);
      float ps = p0 + p1;
      ps += __shfl_xor(ps, 1, 32);
      ps += __shfl_xor(ps, 2, 32);
      ps += __shfl_xor(ps, 4, 32);
      ps += __shfl_xor(ps, 8, 32);
      srow[i] = srow[i] * alpha + ps;
#pragma unroll
      for (int c = 0; c < 8; ++c) O[c][i] *= alpha;
      tile[wv][hi * 8 + i][ln]      = (f16)p0;
      tile[wv][hi * 8 + i][16 + ln] = (f16)p1;
    }
    asm volatile("s_wait_dscnt 0" ::: "memory");
    v16h pa;
    {
      H16 r;
      r.u[0] = *(const uint4*)&tile[wv][ln][hi * 16];
      r.u[1] = *(const uint4*)&tile[wv][ln][hi * 16 + 8];
      pa = r.v;
    }
#pragma unroll
    for (int c = 0; c < 8; ++c) {
      v16h vb = load16(vbase + (size_t)(c * 16 + ln) * SEQ + n0 + hi * 16);
      O[c] = wmma16(pa, vb, O[c]);
    }
  }
#pragma unroll
  for (int i = 0; i < 8; ++i) {
    float inv = 1.0f / srow[i];
    size_t orow = (((size_t)(b * 16 + sh)) * SEQ + q0 + hi * 8 + i) * 128;
#pragma unroll
    for (int c = 0; c < 8; ++c)
      Obuf[orow + c * 16 + ln] = O[c][i] * inv;
  }
}

// ---------------- differential combine + LayerNorm -> (b,t,C) f16 ----------------
__global__ void k_combine(const float* __restrict__ Obuf, const float* __restrict__ lamp,
                          f16* __restrict__ Yh) {
  const int lane = threadIdx.x & 31;
  const int wv   = threadIdx.x >> 5;
  const int r = blockIdx.x * 8 + wv;      // (b*8+h)*T + t
  const int t  = r & (SEQ - 1);
  const int bh = r >> 11;
  const int h  = bh & 7;
  const int b  = bh >> 3;
  const float lam = lamp[0];
  const float* o1 = Obuf + (((size_t)(b * 16 + 2 * h)) * SEQ + t) * 128;
  const float* o2 = Obuf + (((size_t)(b * 16 + 2 * h + 1)) * SEQ + t) * 128;
  float a[4];
  float mu = 0.0f;
#pragma unroll
  for (int j = 0; j < 4; ++j) {
    int d = lane + j * 32;
    a[j] = o1[d] - lam * o2[d];
    mu += a[j];
  }
  for (int m = 16; m >= 1; m >>= 1) mu += __shfl_xor(mu, m, 32);
  mu *= (1.0f / 128.0f);
  float var = 0.0f;
#pragma unroll
  for (int j = 0; j < 4; ++j) { float dd = a[j] - mu; var += dd * dd; }
  for (int m = 16; m >= 1; m >>= 1) var += __shfl_xor(var, m, 32);
  var *= (1.0f / 128.0f);
  float rs = rsqrtf(var + LN_EPS) * (1.0f - LAMBDA_INIT);
  f16* yrow = Yh + ((size_t)(b * SEQ + t)) * EMB + h * 128;
#pragma unroll
  for (int j = 0; j < 4; ++j) yrow[lane + j * 32] = (f16)((a[j] - mu) * rs);
}

extern "C" void kernel_launch(void* const* d_in, const int* in_sizes, int n_in,
                              void* d_out, int out_size, void* d_ws, size_t ws_size,
                              hipStream_t stream) {
  (void)in_sizes; (void)n_in; (void)out_size; (void)ws_size;
  const float* x   = (const float*)d_in[0];
  const float* Wq  = (const float*)d_in[1];
  const float* Wk  = (const float*)d_in[2];
  const float* Wv  = (const float*)d_in[3];
  const float* Wp  = (const float*)d_in[4];
  const float* lq1 = (const float*)d_in[5];
  const float* lk1 = (const float*)d_in[6];
  const float* lq2 = (const float*)d_in[7];
  const float* lk2 = (const float*)d_in[8];

  char* ws = (char*)d_ws;
  const size_t MB = 1024ull * 1024ull;
  f16*   xh   = (f16*)(ws + 0);          // 16 MB
  f16*   WqH  = (f16*)(ws + 16 * MB);    //  2 MB
  f16*   WkH  = (f16*)(ws + 18 * MB);
  f16*   WvH  = (f16*)(ws + 20 * MB);
  f16*   WpH  = (f16*)(ws + 22 * MB);
  float* qraw = (float*)(ws + 24 * MB);  // 32 MB
  float* kraw = (float*)(ws + 56 * MB);  // 32 MB
  float* vraw = (float*)(ws + 88 * MB);  // 32 MB
  f16*   qrot = (f16*)(ws + 120 * MB);   // 16 MB
  f16*   krot = (f16*)(ws + 136 * MB);   // 16 MB
  f16*   vT   = (f16*)(ws + 152 * MB);   // 16 MB
  float* lamp = (float*)(ws + 168 * MB);
  float* Obuf = qraw;                    // 64 MB reuse (qraw+kraw dead after prep)
  f16*   Yh   = (f16*)vraw;              // 16 MB reuse (vraw dead after prep_v)

  const int M = BATCH * SEQ;             // 8192

  k_cast_h<<<(M * EMB + 255) / 256, 256, 0, stream>>>(x, xh, M * EMB);
  k_cast_h<<<(EMB * EMB + 255) / 256, 256, 0, stream>>>(Wq, WqH, EMB * EMB);
  k_cast_h<<<(EMB * EMB + 255) / 256, 256, 0, stream>>>(Wk, WkH, EMB * EMB);
  k_cast_h<<<(EMB * EMB + 255) / 256, 256, 0, stream>>>(Wv, WvH, EMB * EMB);
  k_cast_h<<<(EMB * EMB + 255) / 256, 256, 0, stream>>>(Wp, WpH, EMB * EMB);

  dim3 gg(EMB / 64, M / 256);
  k_gemm_xwT<<<gg, 256, 0, stream>>>(xh, WqH, qraw, M, EMB, EMB);
  k_gemm_xwT<<<gg, 256, 0, stream>>>(xh, WkH, kraw, M, EMB, EMB);
  k_gemm_xwT<<<gg, 256, 0, stream>>>(xh, WvH, vraw, M, EMB, EMB);

  k_prep_qk<<<(M * 16) / 8, 256, 0, stream>>>(qraw, qrot, 0.125f);  // 1/sqrt(64)
  k_prep_qk<<<(M * 16) / 8, 256, 0, stream>>>(kraw, krot, 1.0f);
  k_prep_v<<<(M * EMB) / 256, 256, 0, stream>>>(vraw, vT);
  k_lambda<<<1, 32, 0, stream>>>(lq1, lk1, lq2, lk2, lamp);

  k_flash<<<dim3(SEQ / 128, BATCH * 16), 256, 0, stream>>>(qrot, krot, vT, Obuf);
  k_combine<<<(BATCH * NH * SEQ) / 8, 256, 0, stream>>>(Obuf, lamp, Yh);

  k_gemm_xwT<<<gg, 256, 0, stream>>>(Yh, WpH, (float*)d_out, M, EMB, EMB);
}
